// GraphAttentionLayer_34540126994437
// MI455X (gfx1250) — matrix operand order repaired and checked by reading, not compiled, not run on hardware
//
#include <hip/hip_runtime.h>
#include <hip/hip_bf16.h>

typedef __attribute__((ext_vector_type(16))) _Float16 v16h;
typedef __attribute__((ext_vector_type(8)))  _Float16 v8h;
typedef __attribute__((ext_vector_type(8)))  float    v8f;

#define B_   2
#define N_   2048
#define FIN  64
#define FOUT 128
#define H_   4
#define D_   32

#define CHUNK 256          // j-span staged in LDS per iteration
#define PROW  (CHUNK + 8)  // padded LDS row stride (halves): 132 dwords -> bank step 4

// ---------------------------------------------------------------------------
// Kernel 1: h = x @ w^T   (per-row block of 128 threads), plus attention
// logits ei[b,h,n] = <h[b,n,h,:], a1[h]>, ej[b,h,n] = <h[b,n,h,:], a2[h]>.
// ---------------------------------------------------------------------------
__global__ __launch_bounds__(128) void gat_proj(
    const float* __restrict__ x, const float* __restrict__ w,
    const float* __restrict__ a, float* __restrict__ h_ws,
    float* __restrict__ ei_ws, float* __restrict__ ej_ws) {
  int row = blockIdx.x;           // b*N + n
  int b = row / N_, n = row % N_;
  int t = threadIdx.x;            // 0..127 : one output feature each
  __shared__ float xs[FIN];
  __shared__ float hs[FOUT];
  if (t < FIN) xs[t] = x[(size_t)row * FIN + t];
  __syncthreads();
  const float* wr = w + (size_t)t * FIN;
  float acc = 0.f;
#pragma unroll
  for (int k = 0; k < FIN; ++k) acc += xs[k] * wr[k];
  hs[t] = acc;
  h_ws[(size_t)row * FOUT + t] = acc;
  __syncthreads();
  if (t < 2 * H_) {
    int hh = t % H_;
    int which = t / H_;                       // 0 -> ei (a1), 1 -> ej (a2)
    const float* av = a + hh * (2 * D_) + which * D_;
    const float* hv = hs + hh * D_;
    float s = 0.f;
#pragma unroll
    for (int d = 0; d < D_; ++d) s += hv[d] * av[d];
    float* dst = (which == 0) ? ei_ws : ej_ws;
    dst[((size_t)b * H_ + hh) * N_ + n] = s;
  }
}

// ---------------------------------------------------------------------------
// Kernel 1b: ejmax[b*H+h] = max_j ej[b,h,j].  Since LeakyReLU is monotonic,
// lrelu(ei + ejmax) upper-bounds every masked logit -> valid softmax shift.
// ---------------------------------------------------------------------------
__global__ __launch_bounds__(256) void gat_ejmax(const float* __restrict__ ej_ws,
                                                 float* __restrict__ ejmax) {
  int bh = blockIdx.x;
  int t = threadIdx.x;
  __shared__ float red[256];
  const float* p = ej_ws + (size_t)bh * N_;
  float m = -3.4e38f;
  for (int i = t; i < N_; i += 256) m = fmaxf(m, p[i]);
  red[t] = m;
  __syncthreads();
  for (int s = 128; s > 0; s >>= 1) {
    if (t < s) red[t] = fmaxf(red[t], red[t + s]);
    __syncthreads();
  }
  if (t == 0) ejmax[bh] = red[0];
}

// ---------------------------------------------------------------------------
// Kernel 2: build vt[b][h][d][j] (f16) = v[b,j,h,d] with the reference's
// transpose(0,2,1,3).reshape scramble: f=j*H+h; jj=f%N; hh=f/N.
// ---------------------------------------------------------------------------
__global__ void gat_build_vt(const float* __restrict__ h_ws,
                             _Float16* __restrict__ vt) {
  int idx = blockIdx.x * blockDim.x + threadIdx.x;  // B*H*D*N total
  if (idx >= B_ * H_ * D_ * N_) return;
  int j = idx % N_;
  int d = (idx / N_) % D_;
  int h = (idx / (N_ * D_)) % H_;
  int b = idx / (N_ * D_ * H_);
  int f  = j * H_ + h;
  int jj = f % N_;
  int hh = f / N_;
  vt[idx] = (_Float16)h_ws[((size_t)b * N_ + jj) * FOUT + hh * D_ + d];
}

// ---------------------------------------------------------------------------
// Kernel 3: attention. One wave32 per (b, head, 16-row block), 4 waves/WG.
// Per 256-j chunk:
//   compute phase: p = exp(lrelu(ei+ej) - gm) masked by adj (branchless),
//                  row-sum accumulated, f16-packed, ds_store_b128 to a
//                  private LDS tile [16][PROW].
//   wmma phase:    8 tiles of K=32: A-fragment = two ds_load_b128 (the
//                  16-bit A 16x32 layout is two contiguous 8-runs per lane),
//                  B-fragment = two v16h global loads of vt, 2x
//                  v_wmma_f32_16x16x32_f16.
// Softmax 1/sum is folded into a final per-row scale of the accumulators.
// No barriers: each wave owns its LDS tile; DS ops are in-order per wave.
// ---------------------------------------------------------------------------
__device__ __forceinline__ float lrelu(float e) {
  return (e > 0.f) ? e : 0.2f * e;
}

__global__ __launch_bounds__(128) void gat_attn(
    const int* __restrict__ adj, const float* __restrict__ ei_ws,
    const float* __restrict__ ej_ws, const float* __restrict__ ejmax,
    const _Float16* __restrict__ vt, float* __restrict__ hp_ws) {
  __shared__ _Float16 pls[4][16][PROW];

  int blk  = blockIdx.x;                 // b * (N/16) + rowblock
  int b    = blk / (N_ / 16);
  int n0   = (blk % (N_ / 16)) * 16;
  int head = threadIdx.x >> 5;           // 4 waves, one head each
  int lane = threadIdx.x & 31;
  int m    = lane & 15;                  // row within the 16-row block
  int hi16 = lane >> 4;

  size_t ejbase = ((size_t)b * H_ + head) * N_;
  float  ei_val = ei_ws[ejbase + (n0 + m)];
  float  gm     = lrelu(ei_val + ejmax[b * H_ + head]);  // safe softmax shift
  const int* adjrow = adj + (size_t)(n0 + m) * N_;

  _Float16* prow = &pls[head][m][0];
  float lsum = 0.f;
  v8f acc0 = {}, acc1 = {};
  const _Float16* vbase = vt + (((size_t)b * H_ + head) * D_) * N_;

  for (int c = 0; c < N_; c += CHUNK) {
    // ---- compute phase: this lane covers row m, j in [c+hi16*128, +128) --
    int j0 = c + hi16 * 128;
#pragma unroll 4
    for (int t = 0; t < 16; ++t) {
      int j = j0 + t * 8;
      float4 e0 = *(const float4*)(ej_ws + ejbase + j);
      float4 e1 = *(const float4*)(ej_ws + ejbase + j + 4);
      int4   a0 = *(const int4*)(adjrow + j);
      int4   a1 = *(const int4*)(adjrow + j + 4);
      float q0 = lrelu(ei_val + e0.x) - gm; q0 = (a0.x != 0) ? q0 : -100.f;
      float q1 = lrelu(ei_val + e0.y) - gm; q1 = (a0.y != 0) ? q1 : -100.f;
      float q2 = lrelu(ei_val + e0.z) - gm; q2 = (a0.z != 0) ? q2 : -100.f;
      float q3 = lrelu(ei_val + e0.w) - gm; q3 = (a0.w != 0) ? q3 : -100.f;
      float q4 = lrelu(ei_val + e1.x) - gm; q4 = (a1.x != 0) ? q4 : -100.f;
      float q5 = lrelu(ei_val + e1.y) - gm; q5 = (a1.y != 0) ? q5 : -100.f;
      float q6 = lrelu(ei_val + e1.z) - gm; q6 = (a1.z != 0) ? q6 : -100.f;
      float q7 = lrelu(ei_val + e1.w) - gm; q7 = (a1.w != 0) ? q7 : -100.f;
      float p0 = __expf(q0), p1 = __expf(q1), p2 = __expf(q2), p3 = __expf(q3);
      float p4 = __expf(q4), p5 = __expf(q5), p6 = __expf(q6), p7 = __expf(q7);
      lsum += ((p0 + p1) + (p2 + p3)) + ((p4 + p5) + (p6 + p7));
      v8h pv;
      pv[0] = (_Float16)p0; pv[1] = (_Float16)p1;
      pv[2] = (_Float16)p2; pv[3] = (_Float16)p3;
      pv[4] = (_Float16)p4; pv[5] = (_Float16)p5;
      pv[6] = (_Float16)p6; pv[7] = (_Float16)p7;
      *(v8h*)(prow + hi16 * 128 + t * 8) = pv;   // 16B-aligned ds_store_b128
    }

    // ---- wmma phase: 8 tiles of K=32 from this chunk ---------------------
#pragma unroll
    for (int t8 = 0; t8 < 8; ++t8) {
      int jl = t8 * 32;                       // j within chunk
      // A-fragment: lane<16 -> K {0..15}; lane>=16 -> K {16..31}
      v8h alo = *(const v8h*)(prow + jl + hi16 * 16);
      v8h ahi = *(const v8h*)(prow + jl + hi16 * 16 + 8);
      v16h af;
#pragma unroll
      for (int k = 0; k < 8; ++k) { af[k] = alo[k]; af[k + 8] = ahi[k]; }
      int jg = c + jl + hi16 * 16;
      v16h bf0 = *(const v16h*)(vbase + (size_t)m * N_        + jg);
      v16h bf1 = *(const v16h*)(vbase + (size_t)(m + 16) * N_ + jg);
      acc0 = __builtin_amdgcn_wmma_f32_16x16x32_f16(false, af, false, bf0,
                                                    (short)0, acc0, false, false);
      acc1 = __builtin_amdgcn_wmma_f32_16x16x32_f16(false, af, false, bf1,
                                                    (short)0, acc1, false, false);
    }
  }

  // ---- fold softmax normalizer into accumulators, store ------------------
  float gs  = lsum + __shfl_xor(lsum, 16, 32);
  float inv = (gs > 0.f) ? __builtin_amdgcn_rcpf(gs) : 0.f;
#pragma unroll
  for (int r = 0; r < 8; ++r) {
    int M = r + hi16 * 8;
    float invr = __shfl(inv, M, 32);          // inv of row M lives on lane M
    size_t o = ((size_t)b * N_ + n0 + M) * FOUT + head * D_;
    hp_ws[o + m]      = acc0[r] * invr;
    hp_ws[o + 16 + m] = acc1[r] * invr;
  }
}

// ---------------------------------------------------------------------------
// Kernel 4: y = h' + h, LayerNorm over 128 features, scale/shift.
// ---------------------------------------------------------------------------
__global__ __launch_bounds__(128) void gat_norm(
    const float* __restrict__ h_ws, const float* __restrict__ hp_ws,
    const float* __restrict__ gamma, const float* __restrict__ beta,
    float* __restrict__ out) {
  int row = blockIdx.x;
  int t = threadIdx.x;
  __shared__ float red[128];
  size_t o = (size_t)row * FOUT + t;
  float y = h_ws[o] + hp_ws[o];
  red[t] = y;
  __syncthreads();
  for (int s = 64; s > 0; s >>= 1) { if (t < s) red[t] += red[t + s]; __syncthreads(); }
  float mu = red[0] * (1.0f / FOUT);
  __syncthreads();
  float d = y - mu;
  red[t] = d * d;
  __syncthreads();
  for (int s = 64; s > 0; s >>= 1) { if (t < s) red[t] += red[t + s]; __syncthreads(); }
  float var = red[0] * (1.0f / FOUT);
  out[o] = d * rsqrtf(var + 1e-5f) * gamma[t] + beta[t];
}

// ---------------------------------------------------------------------------
// Workspace layout (bytes):
//   [0, 2MiB)            h_ws   : B*N*128 f32
//   [2MiB, 4MiB)         hp_ws  : B*N*128 f32
//   [4MiB, +64KiB)       ei_ws  : B*H*N f32
//   [+64KiB, +128KiB)    ej_ws  : B*H*N f32
//   [4MiB+128KiB, +1MiB) vt     : B*H*D*N f16 (32B-aligned)
//   [5MiB+128KiB, +32B)  ejmax  : B*H f32
// Total ~5.25 MiB.
// ---------------------------------------------------------------------------
extern "C" void kernel_launch(void* const* d_in, const int* in_sizes, int n_in,
                              void* d_out, int out_size, void* d_ws, size_t ws_size,
                              hipStream_t stream) {
  const float* x     = (const float*)d_in[0];
  const int*   adj   = (const int*)d_in[1];
  const float* w     = (const float*)d_in[2];
  const float* a     = (const float*)d_in[3];
  const float* gamma = (const float*)d_in[4];
  const float* beta  = (const float*)d_in[5];
  float* out = (float*)d_out;

  char* ws = (char*)d_ws;
  float*    h_ws  = (float*)(ws);
  float*    hp_ws = (float*)(ws + (size_t)(2u << 20));
  float*    ei_ws = (float*)(ws + (size_t)(4u << 20));
  float*    ej_ws = (float*)(ws + (size_t)(4u << 20) + (64u << 10));
  _Float16* vt    = (_Float16*)(ws + (size_t)(4u << 20) + (128u << 10));
  float*    ejm   = (float*)(ws + (size_t)(5u << 20) + (128u << 10));

  gat_proj<<<B_ * N_, 128, 0, stream>>>(x, w, a, h_ws, ei_ws, ej_ws);
  gat_ejmax<<<B_ * H_, 256, 0, stream>>>(ej_ws, ejm);
  gat_build_vt<<<(B_ * H_ * D_ * N_ + 255) / 256, 256, 0, stream>>>(h_ws, vt);
  gat_attn<<<B_ * (N_ / 16), 128, 0, stream>>>(adj, ei_ws, ej_ws, ejm, vt, hp_ws);
  gat_norm<<<B_ * N_, 128, 0, stream>>>(h_ws, hp_ws, gamma, beta, out);
}
